// MoE_31593779429526
// MI455X (gfx1250) — compile-verified, matrix operand order checked by reading, not run on hardware
//
#include <hip/hip_runtime.h>
#include <hip/hip_bf16.h>

// MoE top-2 SwiGLU forward for MI455X (gfx1250, wave32, WMMA).
// Grouped-expert bf16 WMMA pipeline with LDS-staged (transposed+converted)
// weight tiles so all global traffic is b128 and B fragments are ds_load_b128.

typedef __attribute__((ext_vector_type(16))) __bf16 v16bf;
typedef __attribute__((ext_vector_type(8)))  float  v8f;

#define EXPERTS 8
#define NTOPK   2
#define LDS_STRIDE 20   // dwords per column: 16 K-pairs + 4 pad; 80B keeps 16B alignment

union BF16x2 { unsigned int u; __bf16 h[2]; };
union ABReg  { v16bf v; unsigned int u[8]; };

// ---------------------------------------------------------------- router ----
__global__ void router_topk(const float* __restrict__ x, const float* __restrict__ Wg,
                            int* __restrict__ idx_out, float* __restrict__ prob_out,
                            int N, int C) {
    int wid  = threadIdx.x >> 5;
    int lane = threadIdx.x & 31;
    int n = blockIdx.x * 8 + wid;
    if (n >= N) return;

    float acc[EXPERTS];
    #pragma unroll
    for (int e = 0; e < EXPERTS; ++e) acc[e] = 0.f;

    const float* xr = x + (size_t)n * C;
    for (int c = lane; c < C; c += 32) {
        float xv = xr[c];
        const float* wgr = Wg + (size_t)c * EXPERTS;
        #pragma unroll
        for (int e = 0; e < EXPERTS; ++e) acc[e] += xv * wgr[e];
    }
    #pragma unroll
    for (int e = 0; e < EXPERTS; ++e) {
        #pragma unroll
        for (int ofs = 16; ofs > 0; ofs >>= 1)
            acc[e] += __shfl_xor(acc[e], ofs, 32);
    }
    if (lane == 0) {
        // top-2, ties -> lower index (matches jax.lax.top_k)
        int b0 = 0; float v0 = acc[0];
        #pragma unroll
        for (int e = 1; e < EXPERTS; ++e) if (acc[e] > v0) { v0 = acc[e]; b0 = e; }
        int b1 = -1; float v1 = -__builtin_inff();
        #pragma unroll
        for (int e = 0; e < EXPERTS; ++e)
            if (e != b0 && acc[e] > v1) { v1 = acc[e]; b1 = e; }
        float e1 = __expf(v1 - v0);          // v0 >= v1, stable softmax
        float p0 = 1.f / (1.f + e1);
        float p1 = e1 * p0;
        idx_out[n * 2 + 0] = b0;
        idx_out[n * 2 + 1] = b1;
        prob_out[n * 2 + 0] = p0;
        prob_out[n * 2 + 1] = p1;
    }
}

// ----------------------------------------------------------- small utils ----
__global__ void cvt_x_bf16(const float* __restrict__ x, __bf16* __restrict__ xb, long total) {
    // 4 floats -> 2 packed bf16 dwords per iteration
    long i4 = (long)blockIdx.x * blockDim.x + threadIdx.x;
    long stride = (long)gridDim.x * blockDim.x;
    long n4 = total >> 2;
    unsigned int* out = (unsigned int*)xb;
    const float4* in = (const float4*)x;
    for (; i4 < n4; i4 += stride) {
        float4 v = in[i4];
        BF16x2 p0, p1;
        p0.h[0] = (__bf16)v.x; p0.h[1] = (__bf16)v.y;
        p1.h[0] = (__bf16)v.z; p1.h[1] = (__bf16)v.w;
        out[i4 * 2 + 0] = p0.u;
        out[i4 * 2 + 1] = p1.u;
    }
}

__global__ void zero_out(float* __restrict__ y, long total) {
    long i = (long)blockIdx.x * blockDim.x + threadIdx.x;
    long stride = (long)gridDim.x * blockDim.x;
    for (; i < total; i += stride) y[i] = 0.f;
}

// Deterministic compaction: one wave per expert, ballot-prefix over tokens
// in ascending token order -> slot order is reproducible every call.
__global__ void assign_slots(const int* __restrict__ idx, const float* __restrict__ prob,
                             int* __restrict__ tok_of, float* __restrict__ wt_of,
                             int* __restrict__ cnt, int* __restrict__ basep, int N) {
    __shared__ int scnt[EXPERTS];
    int wid  = threadIdx.x >> 5;
    int lane = threadIdx.x & 31;
    if (wid < EXPERTS) {
        int e = wid;
        int count = 0;
        for (int b = 0; b < N; b += 32) {
            int n = b + lane;
            bool m0 = false, m1 = false;
            if (n < N) {
                m0 = (idx[n * 2 + 0] == e);
                m1 = (idx[n * 2 + 1] == e);
            }
            bool m = m0 || m1;
            unsigned int bal = (unsigned int)__ballot(m);
            int slot = count + __popc(bal & ((1u << lane) - 1u));
            if (m) {
                tok_of[e * N + slot] = n;
                wt_of[e * N + slot] = m0 ? prob[n * 2 + 0] : prob[n * 2 + 1];
            }
            count += __popc(bal);
        }
        if (lane == 0) { cnt[e] = count; scnt[e] = count; }
    }
    __syncthreads();
    if (threadIdx.x == 0) {
        int acc = 0;
        for (int e = 0; e < EXPERTS; ++e) { basep[e] = acc; acc += scnt[e]; }
    }
}

// --------------------------------------------------------- GEMM fragments ----
// Stage one 32(K) x 64(N) fp32 weight tile into LDS as bf16 K-pairs:
// lds[col * LDS_STRIDE + kpair] = pack(bf16 W[k, col], bf16 W[k+1, col]).
// Global side: coalesced b128 row-segment loads; conversion via v_cvt_pk_bf16.
__device__ __forceinline__ void stage_w_tile(const float* __restrict__ w, int ldw,
                                             int kc, int nbase,
                                             unsigned int* __restrict__ lds) {
    int t  = threadIdx.x;
    int kp = t >> 4;            // 0..15 -> rows (kc+2kp, kc+2kp+1)
    int n0 = (t & 15) * 4;      // 0..60
    const float* r0 = w + (size_t)(kc + 2 * kp) * ldw + nbase + n0;
    const float* r1 = r0 + ldw;
    float4 a = *(const float4*)r0;
    float4 b = *(const float4*)r1;
    #pragma unroll
    for (int j = 0; j < 4; ++j) {
        BF16x2 p;
        p.h[0] = (__bf16)((&a.x)[j]);
        p.h[1] = (__bf16)((&b.x)[j]);
        lds[(n0 + j) * LDS_STRIDE + kp] = p.u;
    }
}

// B fragment for one 32x16 bf16 tile: two aligned ds_load_b128 runs.
// K-pair map (ISA 7.12.2): dword j<4 -> kp = hi*4+j ; j>=4 -> kp = 8+hi*4+(j-4).
__device__ __forceinline__ v16bf load_b_frag(const unsigned int* __restrict__ lds,
                                             int col, int hi) {
    ABReg b;
    *(uint4*)&b.u[0] = *(const uint4*)(lds + col * LDS_STRIDE + hi * 4);
    *(uint4*)&b.u[4] = *(const uint4*)(lds + col * LDS_STRIDE + 8 + hi * 4);
    return b.v;
}

// A fragment (bf16 rows, K contiguous): two global b128 loads.
__device__ __forceinline__ v16bf load_a_frag(const __bf16* __restrict__ arow,
                                             int kc, int hi) {
    ABReg a;
    *(uint4*)&a.u[0] = *(const uint4*)(arow + kc + hi * 8);
    *(uint4*)&a.u[4] = *(const uint4*)(arow + kc + 16 + hi * 8);
    return a.v;
}

// ---------------------------------------------------- GEMM1: h = silu(xW1)*xW3
// Block: 256 thr = 8 waves as 4(M) x 2(N) -> 64x64 tile; K over C in 32-steps.
__global__ __launch_bounds__(256)
void expert_ffn1(const __bf16* __restrict__ xb,
                 const float* __restrict__ W1, const float* __restrict__ W3,
                 const int* __restrict__ tok_of, const int* __restrict__ cnt,
                 const int* __restrict__ basep, __bf16* __restrict__ H,
                 int N, int C, int F) {
    __shared__ unsigned int lb1[64 * LDS_STRIDE];
    __shared__ unsigned int lb3[64 * LDS_STRIDE];

    const int mtiles = (N + 63) / 64;
    int e  = blockIdx.x / mtiles;
    int mt = blockIdx.x % mtiles;
    int ce = cnt[e];
    if (mt * 64 >= ce) return;

    int wid  = threadIdx.x >> 5;
    int lane = threadIdx.x & 31;
    int wm = wid >> 1, wn = wid & 1;
    int hi = lane >> 4, ln = lane & 15;

    int slotA = mt * 64 + wm * 16 + ln;
    int tokA  = tok_of[e * N + (slotA < ce ? slotA : 0)];
    const __bf16* arow = xb + (size_t)tokA * C;

    const float* w1e = W1 + (size_t)e * C * F;
    const float* w3e = W3 + (size_t)e * C * F;
    int fbase = blockIdx.y * 64;
    int col0 = wn * 32 + ln;          // column within 64-wide tile, sub 0
    int col1 = col0 + 16;             // sub 1

    v8f acc10 = {}, acc11 = {}, acc30 = {}, acc31 = {};

    for (int kc = 0; kc < C; kc += 32) {
        stage_w_tile(w1e, F, kc, fbase, lb1);
        stage_w_tile(w3e, F, kc, fbase, lb3);
        if (kc + 32 < C) { // warm next K-chunk (global_prefetch_b8)
            __builtin_prefetch(&w1e[(size_t)(kc + 32 + (threadIdx.x >> 4) * 2) * F + fbase], 0, 1);
            __builtin_prefetch(&w3e[(size_t)(kc + 32 + (threadIdx.x >> 4) * 2) * F + fbase], 0, 1);
        }
        __syncthreads();

        v16bf a = load_a_frag(arow, kc, hi);
        v16bf b;
        b = load_b_frag(lb1, col0, hi);
        acc10 = __builtin_amdgcn_wmma_f32_16x16x32_bf16(false, a, false, b, (short)0, acc10, false, false);
        b = load_b_frag(lb1, col1, hi);
        acc11 = __builtin_amdgcn_wmma_f32_16x16x32_bf16(false, a, false, b, (short)0, acc11, false, false);
        b = load_b_frag(lb3, col0, hi);
        acc30 = __builtin_amdgcn_wmma_f32_16x16x32_bf16(false, a, false, b, (short)0, acc30, false, false);
        b = load_b_frag(lb3, col1, hi);
        acc31 = __builtin_amdgcn_wmma_f32_16x16x32_bf16(false, a, false, b, (short)0, acc31, false, false);

        __syncthreads();
    }

    int gb = basep[e];
    int fc0 = fbase + col0, fc1 = fbase + col1;
    #pragma unroll
    for (int r = 0; r < 8; ++r) {
        int slot = mt * 64 + wm * 16 + r + hi * 8;  // C/D layout: M = r + 8*hi, N = lane&15
        if (slot < ce) {
            size_t row = (size_t)(gb + slot) * F;
            float h1 = acc10[r], h3 = acc30[r];
            H[row + fc0] = (__bf16)((h1 / (1.f + __expf(-h1))) * h3);
            h1 = acc11[r]; h3 = acc31[r];
            H[row + fc1] = (__bf16)((h1 / (1.f + __expf(-h1))) * h3);
        }
    }
}

// -------------------------------------------- GEMM2: y += wt * (H @ W2[e]) --
__global__ __launch_bounds__(256)
void expert_ffn2(const __bf16* __restrict__ H, const float* __restrict__ W2,
                 const int* __restrict__ tok_of, const float* __restrict__ wt_of,
                 const int* __restrict__ cnt, const int* __restrict__ basep,
                 float* __restrict__ y, int N, int C, int F) {
    __shared__ unsigned int lb2[64 * LDS_STRIDE];

    const int mtiles = (N + 63) / 64;
    int e  = blockIdx.x / mtiles;
    int mt = blockIdx.x % mtiles;
    int ce = cnt[e];
    if (mt * 64 >= ce) return;

    int wid  = threadIdx.x >> 5;
    int lane = threadIdx.x & 31;
    int wm = wid >> 1, wn = wid & 1;
    int hi = lane >> 4, ln = lane & 15;

    int gb = basep[e];
    int slotA = mt * 64 + wm * 16 + ln;
    const __bf16* arow = H + (size_t)(gb + (slotA < ce ? slotA : 0)) * F;
    const float* w2e = W2 + (size_t)e * F * C;
    int cbase = blockIdx.y * 64;
    int col0 = wn * 32 + ln, col1 = col0 + 16;

    v8f acc0 = {}, acc1 = {};
    for (int kc = 0; kc < F; kc += 32) {
        stage_w_tile(w2e, C, kc, cbase, lb2);
        if (kc + 32 < F)
            __builtin_prefetch(&w2e[(size_t)(kc + 32 + (threadIdx.x >> 4) * 2) * C + cbase], 0, 1);
        __syncthreads();

        v16bf a = load_a_frag(arow, kc, hi);
        v16bf b;
        b = load_b_frag(lb2, col0, hi);
        acc0 = __builtin_amdgcn_wmma_f32_16x16x32_bf16(false, a, false, b, (short)0, acc0, false, false);
        b = load_b_frag(lb2, col1, hi);
        acc1 = __builtin_amdgcn_wmma_f32_16x16x32_bf16(false, a, false, b, (short)0, acc1, false, false);

        __syncthreads();
    }

    int cc0 = cbase + col0, cc1 = cbase + col1;
    #pragma unroll
    for (int r = 0; r < 8; ++r) {
        int slot = mt * 64 + wm * 16 + r + hi * 8;
        if (slot < ce) {
            int tok  = tok_of[e * N + slot];
            float wt = wt_of[e * N + slot];
            // exactly 2 contributions per y element; fp add commutes -> deterministic
            atomicAdd(&y[(size_t)tok * C + cc0], wt * acc0[r]);
            atomicAdd(&y[(size_t)tok * C + cc1], wt * acc1[r]);
        }
    }
}

// ------------------------------------------------------------------ launch --
extern "C" void kernel_launch(void* const* d_in, const int* in_sizes, int n_in,
                              void* d_out, int out_size, void* d_ws, size_t ws_size,
                              hipStream_t stream) {
    const float* x  = (const float*)d_in[0];
    const float* Wg = (const float*)d_in[1];
    const float* W1 = (const float*)d_in[2];
    const float* W3 = (const float*)d_in[3];
    const float* W2 = (const float*)d_in[4];

    const int E = EXPERTS;
    const int C = in_sizes[1] / E;                          // 1024
    const int N = (int)((long)in_sizes[0] / C);             // 2048
    const int F = (int)((long)in_sizes[2] / ((long)E * C)); // 2816

    float* y = (float*)d_out;
    int* idx_out = (int*)((float*)d_out + (size_t)N * C);   // indices tail, int32 bits

    // workspace carve-out (~28 MB total)
    char* ws = (char*)d_ws;
    size_t off = 0;
    auto carve = [&](size_t bytes) -> void* {
        void* p = ws + off;
        off = (off + bytes + 255) & ~(size_t)255;
        return p;
    };
    __bf16* xb    = (__bf16*)carve((size_t)N * C * sizeof(__bf16));         // 4 MB
    float*  prob  = (float*) carve((size_t)N * NTOPK * sizeof(float));
    int*    tok_of= (int*)   carve((size_t)E * N * sizeof(int));
    float*  wt_of = (float*) carve((size_t)E * N * sizeof(float));
    int*    cnt   = (int*)   carve(E * sizeof(int));
    int*    basep = (int*)   carve(E * sizeof(int));
    __bf16* H     = (__bf16*)carve((size_t)N * NTOPK * F * sizeof(__bf16)); // 23 MB
    (void)ws_size; (void)n_in; (void)out_size;

    router_topk<<<(N + 7) / 8, 256, 0, stream>>>(x, Wg, idx_out, prob, N, C);
    cvt_x_bf16 <<<512, 256, 0, stream>>>(x, xb, (long)N * C);
    zero_out   <<<512, 256, 0, stream>>>(y, (long)N * C);
    assign_slots<<<1, 256, 0, stream>>>(idx_out, prob, tok_of, wt_of, cnt, basep, N);

    int mtiles = (N + 63) / 64;
    dim3 g1(E * mtiles, F / 64);
    expert_ffn1<<<g1, 256, 0, stream>>>(xb, W1, W3, tok_of, cnt, basep, H, N, C, F);
    dim3 g2(E * mtiles, C / 64);
    expert_ffn2<<<g2, 256, 0, stream>>>(H, W2, tok_of, wt_of, cnt, basep, y, N, C, F);
}
